// CausalSelfAttention_26328149524651
// MI455X (gfx1250) — matrix operand order repaired and checked by reading, not compiled, not run on hardware
//
#include <hip/hip_runtime.h>

typedef __bf16 bf16;
typedef bf16 v16bf __attribute__((ext_vector_type(16)));
typedef float v8f __attribute__((ext_vector_type(8)));
typedef int i32x4 __attribute__((__vector_size__(16)));

#define GAS __attribute__((address_space(1)))
#define LAS __attribute__((address_space(3)))

#if __has_builtin(__builtin_amdgcn_global_load_async_to_lds_b128)
#define HAVE_ASYNC_LDS 1
#else
#define HAVE_ASYNC_LDS 0
#endif
#if __has_builtin(__builtin_amdgcn_s_wait_asynccnt)
#define HAVE_WAIT_ASYNC 1
#else
#define HAVE_WAIT_ASYNC 0
#endif

// Async global->LDS copy, 16B per lane (ASYNCcnt). Falls back to VGPR copy.
__device__ __forceinline__ void async_b128(void* lds, const void* g) {
#if HAVE_ASYNC_LDS
  __builtin_amdgcn_global_load_async_to_lds_b128((GAS i32x4*)g, (LAS i32x4*)lds,
                                                 0, 0);
#else
  *(uint4*)lds = *(const uint4*)g;
#endif
}

template <int N>
__device__ __forceinline__ void wait_async() {
#if HAVE_ASYNC_LDS
#if HAVE_WAIT_ASYNC
  __builtin_amdgcn_s_wait_asynccnt(N);
#else
  asm volatile("s_wait_asynccnt %0" ::"i"(N) : "memory");
#endif
#endif
}

union ABFrag { v16bf v; unsigned int u[8]; };

__device__ __forceinline__ v8f wmma_bf16(const ABFrag& a, const ABFrag& b, v8f c) {
  // D = A(16x32 bf16) * B(32x16 bf16) + C(16x16 f32)
  return __builtin_amdgcn_wmma_f32_16x16x32_bf16(false, a.v, false, b.v,
                                                 (short)0, c, false, false);
}

// A operand, contiguous k order used here (not ISA table order): lane%16 = M row, VGPR v holds K = 2v + 16*half.
__device__ __forceinline__ int kAoff(int v, int h) {
  return v * 2 + h * 16;
}
// B operand, contiguous k order used here (not ISA table order): lane%16 = N col, VGPR v holds K = 2v + 16*half.
__device__ __forceinline__ int kBoff(int v, int h) { return v * 2 + h * 16; }

__device__ __forceinline__ ABFrag load_afrag(const bf16* base, int stride,
                                             int row, int k0, int h) {
  ABFrag f;
  const bf16* p = base + row * stride + k0;
#pragma unroll
  for (int v = 0; v < 8; ++v)
    f.u[v] = *(const unsigned int*)(p + kAoff(v, h));
  return f;
}
__device__ __forceinline__ ABFrag load_bfrag(const bf16* base, int stride,
                                             int col, int k0, int h) {
  ABFrag f;
  const bf16* p = base + col * stride + k0;
#pragma unroll
  for (int v = 0; v < 8; ++v)
    f.u[v] = *(const unsigned int*)(p + kBoff(v, h));
  return f;
}

__device__ __forceinline__ float rowmax16(float x) {
#pragma unroll
  for (int m = 1; m < 16; m <<= 1) x = fmaxf(x, __shfl_xor(x, m, 32));
  return x;
}
__device__ __forceinline__ float rowsum16(float x) {
#pragma unroll
  for (int m = 1; m < 16; m <<= 1) x += __shfl_xor(x, m, 32);
  return x;
}

// ---------------------------------------------------------------------------
// fp32 -> bf16 convert
// ---------------------------------------------------------------------------
__global__ __launch_bounds__(256) void cvt_f32_bf16(const float* __restrict__ x,
                                                    bf16* __restrict__ o) {
  long i = ((long)blockIdx.x * 256 + threadIdx.x) * 4;
  float4 v = *(const float4*)(x + i);
  union { unsigned long long q; bf16 e[4]; } r;
  r.e[0] = (bf16)v.x; r.e[1] = (bf16)v.y; r.e[2] = (bf16)v.z; r.e[3] = (bf16)v.w;
  *(unsigned long long*)(o + i) = r.q;
}

// ---------------------------------------------------------------------------
// GEMM: out[m][n] = sum_k A[m][k] * W[n][k] + bias[n]   (A bf16, W fp32)
// Block 256(M) x 128(N) x 32(K); 8 waves, wave tile 64x64 (4x4 WMMA tiles).
// A staged via double-buffered async-to-LDS; W converted fp32->bf16 in regs.
// MODE 0: scatter to Q(scaled)/K/V bf16 [B,H,T,D];  MODE 1: fp32 out [M][N]
// ---------------------------------------------------------------------------
template <int MODE>
__global__ __launch_bounds__(256) void gemm_bf16_kernel(
    const bf16* __restrict__ A, const float* __restrict__ W,
    const float* __restrict__ bias, int K, int N, float* __restrict__ outF,
    bf16* __restrict__ Qo, bf16* __restrict__ Ko, bf16* __restrict__ Vo) {
  __shared__ bf16 As[2][256][32];
  __shared__ bf16 Ws[128][32];

  const int tid = threadIdx.x;
  const int lane = tid & 31, wave = tid >> 5;
  const int h = lane >> 4, l16 = lane & 15;
  const int wm = wave & 3, wn = wave >> 2;  // wave tile: rows wm*64, cols wn*64
  const int m0 = blockIdx.y * 256, n0 = blockIdx.x * 128;

  v8f acc[4][4];
#pragma unroll
  for (int i = 0; i < 4; ++i)
#pragma unroll
    for (int j = 0; j < 4; ++j)
#pragma unroll
      for (int v = 0; v < 8; ++v) acc[i][j][v] = 0.0f;

  const bf16* aRow = A + (long)(m0 + tid) * K;  // thread stages A row `tid`
  const int wr = tid >> 1, wc = (tid & 1) * 16; // W staging: 16 floats/thread
  const float* wRow = W + (long)(n0 + wr) * K;

  // Prologue: stage A tile 0 into buffer 0 (4 x 16B per thread).
#pragma unroll
  for (int c = 0; c < 4; ++c) async_b128(&As[0][tid][c * 8], aRow + c * 8);

  const int nk = K / 32;
  for (int it = 0; it < nk; ++it) {
    const int kt = it * 32;
    const int cur = it & 1;
    const bool more = (kt + 32 < K);

    float4 w0 = *(const float4*)(wRow + kt + wc);
    float4 w1 = *(const float4*)(wRow + kt + wc + 4);
    float4 w2 = *(const float4*)(wRow + kt + wc + 8);
    float4 w3 = *(const float4*)(wRow + kt + wc + 12);
    if (more) __builtin_prefetch(wRow + kt + 32 + wc, 0, 1);

    __syncthreads();  // previous iteration's readers of As[cur^1] / Ws done

    if (more) {  // async-stage next A tile while we compute this one
#pragma unroll
      for (int c = 0; c < 4; ++c)
        async_b128(&As[cur ^ 1][tid][c * 8], aRow + kt + 32 + c * 8);
    }
    union { uint4 q[2]; bf16 e[16]; } wt;
    wt.e[0] = (bf16)w0.x;  wt.e[1] = (bf16)w0.y;
    wt.e[2] = (bf16)w0.z;  wt.e[3] = (bf16)w0.w;
    wt.e[4] = (bf16)w1.x;  wt.e[5] = (bf16)w1.y;
    wt.e[6] = (bf16)w1.z;  wt.e[7] = (bf16)w1.w;
    wt.e[8] = (bf16)w2.x;  wt.e[9] = (bf16)w2.y;
    wt.e[10] = (bf16)w2.z; wt.e[11] = (bf16)w2.w;
    wt.e[12] = (bf16)w3.x; wt.e[13] = (bf16)w3.y;
    wt.e[14] = (bf16)w3.z; wt.e[15] = (bf16)w3.w;
    *(uint4*)&Ws[wr][wc] = wt.q[0];
    *(uint4*)&Ws[wr][wc + 8] = wt.q[1];

    if (more) wait_async<4>();  // current tile's 4 async ops complete
    else      wait_async<0>();
    __syncthreads();

    ABFrag af[4], bfr[4];
#pragma unroll
    for (int mi = 0; mi < 4; ++mi)
      af[mi] = load_afrag(&As[cur][0][0], 32, wm * 64 + mi * 16 + l16, 0, h);
#pragma unroll
    for (int ni = 0; ni < 4; ++ni)
      bfr[ni] = load_bfrag(&Ws[0][0], 32, wn * 64 + ni * 16 + l16, 0, h);
#pragma unroll
    for (int mi = 0; mi < 4; ++mi)
#pragma unroll
      for (int ni = 0; ni < 4; ++ni)
        acc[mi][ni] = wmma_bf16(af[mi], bfr[ni], acc[mi][ni]);
  }

  // Epilogue. C layout: VGPR v -> row v + 8*half, col = lane%16.
#pragma unroll
  for (int mi = 0; mi < 4; ++mi)
#pragma unroll
    for (int ni = 0; ni < 4; ++ni)
#pragma unroll
      for (int v = 0; v < 8; ++v) {
        int m = m0 + wm * 64 + mi * 16 + v + 8 * h;
        int n = n0 + wn * 64 + ni * 16 + l16;
        float val = acc[mi][ni][v] + bias[n];
        if (MODE == 0) {
          int which = n >> 10;          // 0=q 1=k 2=v
          int e = n & 1023;
          int hh = e >> 6, d = e & 63;  // head, dim
          int bb = m >> 11, tt = m & 2047;
          long idx = (((long)(bb * 16 + hh)) * 2048 + tt) * 64 + d;
          if (which == 0)      Qo[idx] = (bf16)(val * 0.125f);  // fold 1/sqrt(64)
          else if (which == 1) Ko[idx] = (bf16)val;
          else                 Vo[idx] = (bf16)val;
        } else {
          outF[(long)m * N + n] = val;
        }
      }
}

// ---------------------------------------------------------------------------
// Flash attention: per (q-block of 64, head, batch). 4 waves, wave = 16 q rows.
// K-blocks of 32 keys: S = Q K^T (2 WMMA / 16x16 tile), online softmax,
// P staged through LDS into A-layout, O += P V (4 WMMA), causal mask.
// Q/K staged via async-to-LDS; V transposed through VGPRs.
// ---------------------------------------------------------------------------
__global__ __launch_bounds__(128) void flash_attn_kernel(
    const bf16* __restrict__ Qg, const bf16* __restrict__ Kg,
    const bf16* __restrict__ Vg, bf16* __restrict__ Y) {
  const int T = 2048, D = 64, Hh = 16, E = 1024;
  const int q0 = blockIdx.x * 64;
  const int hIdx = blockIdx.y, bIdx = blockIdx.z;
  const long base = ((long)(bIdx * Hh + hIdx)) * T * D;
  const bf16* Qb = Qg + base + (long)q0 * D;
  const bf16* Kb = Kg + base;
  const bf16* Vb = Vg + base;

  __shared__ bf16 Qs[64][64];
  __shared__ bf16 Ks[32][64];
  __shared__ bf16 Vts[64][36];    // V transposed [d][key], padded
  __shared__ bf16 Ps[4][16][32];  // per-wave P tile

  const int tid = threadIdx.x;
  const int lane = tid & 31, wave = tid >> 5;
  const int h = lane >> 4, l16 = lane & 15;

  {  // stage Q block 64x64 (async, 4 x 16B per thread)
    int r = tid >> 1, c = (tid & 1) * 32;
#pragma unroll
    for (int j = 0; j < 4; ++j)
      async_b128(&Qs[r][c + j * 8], Qb + r * D + c + j * 8);
  }
  wait_async<0>();
  __syncthreads();

  ABFrag qf[2];
  qf[0] = load_afrag(&Qs[0][0], 64, wave * 16 + l16, 0, h);
  qf[1] = load_afrag(&Qs[0][0], 64, wave * 16 + l16, 32, h);

  v8f o[4];
#pragma unroll
  for (int t = 0; t < 4; ++t)
#pragma unroll
    for (int v = 0; v < 8; ++v) o[t][v] = 0.0f;
  float mstate[8], lstate[8];
#pragma unroll
  for (int v = 0; v < 8; ++v) { mstate[v] = -1e30f; lstate[v] = 0.0f; }

  const int qrow_wave = q0 + wave * 16;
  const int nkb = (q0 + 64) / 32;  // causal: keys up to q0+63
  const int sr = tid >> 2, scol = (tid & 3) * 16;

  for (int kb = 0; kb < nkb; ++kb) {
    const int k0 = kb * 32;
    {  // stage K 32x64 row-major (async, 2 x 16B per thread)
      const bf16* s = Kb + (long)(k0 + sr) * D + scol;
      async_b128(&Ks[sr][scol], s);
      async_b128(&Ks[sr][scol + 8], s + 8);
    }
    {  // stage V transposed: Vts[d][key]
      union { uint4 q[2]; bf16 e[16]; } tv;
      const uint4* s = (const uint4*)(Vb + (long)(k0 + sr) * D + scol);
      tv.q[0] = s[0]; tv.q[1] = s[1];
#pragma unroll
      for (int i = 0; i < 16; ++i) Vts[scol + i][sr] = tv.e[i];
    }
    wait_async<0>();
    __syncthreads();

    // S tiles (16q x 16k), keys [k0, k0+16) and [k0+16, k0+32)
    v8f s0, s1;
#pragma unroll
    for (int v = 0; v < 8; ++v) { s0[v] = 0.0f; s1[v] = 0.0f; }
    {
      ABFrag b00 = load_bfrag(&Ks[0][0], 64, l16, 0, h);
      ABFrag b01 = load_bfrag(&Ks[0][0], 64, l16, 32, h);
      ABFrag b10 = load_bfrag(&Ks[0][0], 64, 16 + l16, 0, h);
      ABFrag b11 = load_bfrag(&Ks[0][0], 64, 16 + l16, 32, h);
      s0 = wmma_bf16(qf[0], b00, s0);
      s0 = wmma_bf16(qf[1], b01, s0);
      s1 = wmma_bf16(qf[0], b10, s1);
      s1 = wmma_bf16(qf[1], b11, s1);
    }

    if (k0 + 31 > qrow_wave) {  // causal mask for tiles crossing the diagonal
#pragma unroll
      for (int v = 0; v < 8; ++v) {
        int qr = qrow_wave + v + 8 * h;
        if (k0 + l16 > qr) s0[v] = -1e30f;
        if (k0 + 16 + l16 > qr) s1[v] = -1e30f;
      }
    }

    float corr[8];
#pragma unroll
    for (int v = 0; v < 8; ++v) {
      float mx = rowmax16(fmaxf(s0[v], s1[v]));
      float mn = fmaxf(mstate[v], mx);
      corr[v] = __expf(mstate[v] - mn);
      mstate[v] = mn;
      float p0 = __expf(s0[v] - mn);
      float p1 = __expf(s1[v] - mn);
      float rs = rowsum16(p0 + p1);
      lstate[v] = lstate[v] * corr[v] + rs;
      Ps[wave][v + 8 * h][l16] = (bf16)p0;
      Ps[wave][v + 8 * h][16 + l16] = (bf16)p1;
    }
#pragma unroll
    for (int t = 0; t < 4; ++t)
#pragma unroll
      for (int v = 0; v < 8; ++v) o[t][v] *= corr[v];

    // wave-local: P (16x32) as A fragment, V^T as B fragments
    ABFrag pf = load_afrag(&Ps[wave][0][0], 32, l16, 0, h);
#pragma unroll
    for (int t = 0; t < 4; ++t) {
      ABFrag vf = load_bfrag(&Vts[0][0], 36, t * 16 + l16, 0, h);
      o[t] = wmma_bf16(pf, vf, o[t]);
    }
    __syncthreads();  // before next staging overwrites Ks/Vts
  }

  float inv[8];
#pragma unroll
  for (int v = 0; v < 8; ++v) inv[v] = 1.0f / lstate[v];
#pragma unroll
  for (int t = 0; t < 4; ++t)
#pragma unroll
    for (int v = 0; v < 8; ++v) {
      int qr = q0 + wave * 16 + v + 8 * h;
      int d = t * 16 + l16;
      Y[((long)bIdx * T + qr) * E + hIdx * 64 + d] = (bf16)(o[t][v] * inv[v]);
    }
}

// ---------------------------------------------------------------------------
// launch: B=4 T=2048 E=1024 H=16 D=64
// ws layout: [0,16MB) Xbf (reused as Ybf) | Q | K | V  (16MB each, 64MB total)
// ---------------------------------------------------------------------------
extern "C" void kernel_launch(void* const* d_in, const int* in_sizes, int n_in,
                              void* d_out, int out_size, void* d_ws,
                              size_t ws_size, hipStream_t stream) {
  const float* x = (const float*)d_in[0];
  const float* qkv_w = (const float*)d_in[1];
  const float* qkv_b = (const float*)d_in[2];
  const float* proj_w = (const float*)d_in[3];
  const float* proj_b = (const float*)d_in[4];
  float* out = (float*)d_out;

  const long SEG = 1L << 24;  // 16MB = B*T*E * 2 bytes
  char* ws = (char*)d_ws;
  bf16* Xbf = (bf16*)ws;  // also used as Ybf after attention
  bf16* Qb = (bf16*)(ws + SEG);
  bf16* Kb = (bf16*)(ws + 2 * SEG);
  bf16* Vb = (bf16*)(ws + 3 * SEG);

  // 1) x -> bf16 (8M elems, 4 per thread)
  cvt_f32_bf16<<<8192, 256, 0, stream>>>(x, Xbf);

  // 2) QKV GEMM: M=8192, N=3072, K=1024 -> Q(scaled)/K/V bf16 [B,H,T,D]
  dim3 g1(3072 / 128, 8192 / 256);
  gemm_bf16_kernel<0><<<g1, 256, 0, stream>>>(Xbf, qkv_w, qkv_b, 1024, 3072,
                                              nullptr, Qb, Kb, Vb);

  // 3) Flash attention -> Ybf [B,T,E] (aliases Xbf; GEMM1 consumed it)
  dim3 g2(2048 / 64, 16, 4);
  flash_attn_kernel<<<g2, 128, 0, stream>>>(Qb, Kb, Vb, Xbf);

  // 4) Output projection: M=8192, N=1024, K=1024 -> fp32 out + bias
  dim3 g3(1024 / 128, 8192 / 256);
  gemm_bf16_kernel<1><<<g3, 256, 0, stream>>>(Xbf, proj_w, proj_b, 1024, 1024,
                                              out, nullptr, nullptr, nullptr);
}
